// ChatBlock_27470610825614
// MI455X (gfx1250) — matrix operand order repaired and checked by reading, not compiled, not run on hardware
//
#include <hip/hip_runtime.h>
#include <cstdint>

#define DEV static __device__ __forceinline__

typedef __attribute__((ext_vector_type(16))) __bf16 bf16x16;
typedef __attribute__((ext_vector_type(8)))  __bf16 bf16x8;
typedef __attribute__((ext_vector_type(8)))  float  f32x8;
typedef __attribute__((ext_vector_type(4)))  unsigned int u32x4;
typedef __attribute__((ext_vector_type(8)))  int i32x8;
typedef __attribute__((ext_vector_type(4)))  int i32x4;

#if defined(__HIP_DEVICE_COMPILE__) && __has_builtin(__builtin_amdgcn_tensor_load_to_lds)
#define HAS_TDM 1
#else
#define HAS_TDM 0
#endif

constexpr int kB = 64, kT = 384, kC = 256;
constexpr int kNH = 4, kNKV = 2, kHD = 64;
constexpr int kN = kB * kT;              // 24576 tokens
constexpr int kNE = 8;
constexpr int kHID = 682, kHIDP = 704;   // padded to 22*32

// ---------------- WMMA helpers (gfx1250, wave32) ----------------

DEV f32x8 zero8() { f32x8 z = {0.f,0.f,0.f,0.f,0.f,0.f,0.f,0.f}; return z; }

DEV f32x8 wmma_bf16(bf16x16 a, bf16x16 b, f32x8 c) {
  // D = A(16x32) * B(32x16) + C(16x16), fp32 accumulate
  return __builtin_amdgcn_wmma_f32_16x16x32_bf16(false, a, false, b, (short)0, c, false, false);
}

// A-matrix 16x32 bf16 fragment from row-major [16][stride] tile.
// lanes 0-15: M=lane, K = k0+0..7 and k0+16..23
// lanes 16-31: M=lane-16, K = k0+8..15 and k0+24..31
DEV bf16x16 load_a_frag(const __bf16* base, int stride, int k0, int lane) {
  const int m = lane & 15, hh = (lane >> 4) & 1;
  const __bf16* p = base + (size_t)m * stride + k0 + 8 * hh;
  bf16x8 lo = *(const bf16x8*)(p);
  bf16x8 hi = *(const bf16x8*)(p + 16);
  return __builtin_shufflevector(lo, hi, 0,1,2,3,4,5,6,7,8,9,10,11,12,13,14,15);
}

// B-matrix 32x16 bf16 fragment; `base` points at row-major B^T = W rows
// lanes 0-15 hold K=k0+0..7 and k0+16..23 of column n=lane; lanes 16-31 K=k0+8..15 and k0+24..31
DEV bf16x16 load_b_frag(const __bf16* base, int stride, int k0, int lane) {
  const int n = lane & 15, hh = (lane >> 4) & 1;
  const __bf16* p = base + (size_t)n * stride + k0 + 8 * hh;
  bf16x8 lo = *(const bf16x8*)(p);
  bf16x8 hi = *(const bf16x8*)(p + 16);
  return __builtin_shufflevector(lo, hi, 0,1,2,3,4,5,6,7,8,9,10,11,12,13,14,15);
}

// C/D 16x16 f32 tile element mapping: element r -> (m = r + 8*(lane>>4), n = lane&15)

// ---------------- Prep kernels ----------------

__global__ __launch_bounds__(256) void k_cvt_bf16(__bf16* dst, const float* src, int n) {
  int i = blockIdx.x * 256 + threadIdx.x;
  if (i < n) dst[i] = (__bf16)src[i];
}

// w1/w3: [NE][HID][C] -> padded [NE][HIDP][C] bf16 (zero rows HID..HIDP)
__global__ __launch_bounds__(256) void k_pad_w13(__bf16* dst, const float* src) {
  int i = blockIdx.x * 256 + threadIdx.x;
  const int total = kNE * kHIDP * kC;
  if (i >= total) return;
  int c = i % kC;
  int r = (i / kC) % kHIDP;
  int e = i / (kC * kHIDP);
  dst[i] = (r < kHID) ? (__bf16)src[((size_t)e * kHID + r) * kC + c] : (__bf16)0.f;
}

// w2: [NE][C][HID] -> padded [NE][C][HIDP] bf16 (zero cols HID..HIDP)
__global__ __launch_bounds__(256) void k_pad_w2(__bf16* dst, const float* src) {
  int i = blockIdx.x * 256 + threadIdx.x;
  const int total = kNE * kC * kHIDP;
  if (i >= total) return;
  int h = i % kHIDP;
  int c = (i / kHIDP) % kC;
  int e = i / (kHIDP * kC);
  dst[i] = (h < kHID) ? (__bf16)src[((size_t)e * kC + c) * kHID + h] : (__bf16)0.f;
}

__global__ __launch_bounds__(256) void k_rope_tables(float* rc, float* rs) {
  int i = blockIdx.x * 256 + threadIdx.x;
  if (i >= kT * kHD) return;
  int t = i / kHD, d = i % kHD;
  float invf = powf(10000.f, -(float)(2 * (d & 31)) / (float)kHD);
  float a = (float)t * invf;
  rc[i] = cosf(a);
  rs[i] = sinf(a);
}

__global__ __launch_bounds__(32) void k_zero_cnt(int* cnt) {
  if (threadIdx.x < kNE) cnt[threadIdx.x] = 0;
}

// ---------------- RMSNorm 1 ----------------

__global__ __launch_bounds__(32) void k_rmsnorm1(const float* __restrict__ x,
                                                 const float* __restrict__ w,
                                                 __bf16* __restrict__ outb) {
  int row = blockIdx.x;
  int lane = threadIdx.x & 31;
  const float* xr = x + (size_t)row * kC;
  float ss = 0.f;
  for (int i = lane; i < kC; i += 32) { float v = xr[i]; ss += v * v; }
  for (int off = 16; off; off >>= 1) ss += __shfl_xor(ss, off, 32);
  float r = rsqrtf(ss / (float)kC + 1e-6f);
  for (int i = lane; i < kC; i += 32) outb[(size_t)row * kC + i] = (__bf16)(xr[i] * r * w[i]);
}

// ---------------- QKV projection + RoPE (WMMA) ----------------
// grid (24 t-tiles, 8 slots, B); 1 wave per block.
// slots 0-3: q heads; 4-5: k heads; 6-7: v heads (v written transposed [d][t])

__global__ __launch_bounds__(32) void k_qkv_rope(const __bf16* __restrict__ hb,
                                                 const __bf16* __restrict__ wqb,
                                                 const __bf16* __restrict__ wkb,
                                                 const __bf16* __restrict__ wvb,
                                                 const float* __restrict__ rc,
                                                 const float* __restrict__ rs,
                                                 __bf16* __restrict__ qb,
                                                 __bf16* __restrict__ kb,
                                                 __bf16* __restrict__ vtb) {
  int lane = threadIdx.x & 31;
  int ttile = blockIdx.x, slot = blockIdx.y, b = blockIdx.z;
  int t0 = ttile * 16;
  const __bf16* A = hb + ((size_t)b * kT + t0) * kC;

  const __bf16* W; int head; int typ;
  if (slot < 4)      { W = wqb; head = slot;     typ = 0; }
  else if (slot < 6) { W = wkb; head = slot - 4; typ = 1; }
  else               { W = wvb; head = slot - 6; typ = 2; }
  const int wrow0 = head * kHD;

  f32x8 acc[4] = {zero8(), zero8(), zero8(), zero8()};
  for (int k0 = 0; k0 < kC; k0 += 32) {
    bf16x16 af = load_a_frag(A, kC, k0, lane);
    for (int j = 0; j < 4; ++j) {
      bf16x16 bf = load_b_frag(W + (size_t)(wrow0 + 16 * j) * kC, kC, k0, lane);
      acc[j] = wmma_bf16(af, bf, acc[j]);
    }
  }

  const int n = lane & 15, mo = 8 * (lane >> 4);
  for (int j = 0; j < 4; ++j) {
    for (int r = 0; r < 8; ++r) {
      int m = r + mo;
      int t = t0 + m, d = 16 * j + n;
      float v = acc[j][r];
      if (typ == 2) {
        // V transposed: [b][kv][d][t]
        vtb[(((size_t)b * kNKV + head) * kHD + d) * kT + t] = (__bf16)v;
      } else {
        // RoPE: pair d <-> d +/- 32 lives in acc[j +/- 2], same lane/element
        float rot = (j < 2) ? -acc[j + 2][r] : acc[j - 2][r];
        float cv = rc[t * kHD + d], sv = rs[t * kHD + d];
        float o = v * cv + rot * sv;
        if (typ == 0) qb[(((size_t)b * kNH  + head) * kT + t) * kHD + d] = (__bf16)o;
        else          kb[(((size_t)b * kNKV + head) * kT + t) * kHD + d] = (__bf16)o;
      }
    }
  }
}

// ---------------- Causal attention (WMMA QK^T, softmax, WMMA PV) ----------------
// grid (24 q-tiles, NH, B); 128 threads = 4 waves per block.
// Waves split score tiles (st%4), softmax splits rows x 8 column segments,
// PV phase assigns one 16-wide d-tile per wave.

__global__ __launch_bounds__(128) void k_attn(const __bf16* __restrict__ qb,
                                              const __bf16* __restrict__ kb,
                                              const __bf16* __restrict__ vtb,
                                              __bf16* __restrict__ yb) {
  __shared__ __align__(16) float  sc[16 * kT];
  __shared__ __align__(16) __bf16 pb[16 * 400];
  __shared__ float pmax[8][16];
  __shared__ float psum[8][16];
  __shared__ float rmax[16];
  __shared__ float rinv[16];

  const int tid = threadIdx.x;
  const int w = tid >> 5, lane = tid & 31;
  const int ttile = blockIdx.x, h = blockIdx.y, b = blockIdx.z;
  const int t0 = ttile * 16;
  const int kv = h >> 1;  // GQA: NREP=2

  const __bf16* Q  = qb  + (((size_t)b * kNH  + h)  * kT + t0) * kHD;
  const __bf16* K  = kb  + (((size_t)b * kNKV + kv) * kT) * kHD;
  const __bf16* VT = vtb + (((size_t)b * kNKV + kv) * kHD) * kT;

  bf16x16 qa0 = load_a_frag(Q, kHD, 0,  lane);
  bf16x16 qa1 = load_a_frag(Q, kHD, 32, lane);

  const int n = lane & 15, mo = 8 * (lane >> 4);
  for (int st = w; st <= ttile; st += 4) {
    f32x8 acc = zero8();
    acc = wmma_bf16(qa0, load_b_frag(K + (size_t)st * 16 * kHD, kHD, 0,  lane), acc);
    acc = wmma_bf16(qa1, load_b_frag(K + (size_t)st * 16 * kHD, kHD, 32, lane), acc);
    for (int r = 0; r < 8; ++r) {
      int m = r + mo;
      float s = acc[r] * 0.125f;  // 1/sqrt(64)
      if (st == ttile && n > m) s = -1e30f;
      sc[m * kT + st * 16 + n] = s;
    }
  }
  __syncthreads();

  const int L = (ttile + 1) * 16;
  {
    const int m = tid & 15, seg = tid >> 4;   // 16 rows x 8 column segments
    const int segLen = L >> 3;                // 2*(ttile+1)
    const int i0 = seg * segLen, i1 = i0 + segLen;
    float mx = -1e30f;
    for (int i = i0; i < i1; ++i) mx = fmaxf(mx, sc[m * kT + i]);
    pmax[seg][m] = mx;
    __syncthreads();
    if (tid < 16) {
      float v = pmax[0][tid];
      for (int s = 1; s < 8; ++s) v = fmaxf(v, pmax[s][tid]);
      rmax[tid] = v;
    }
    __syncthreads();
    const float mrow = rmax[m];
    float sum = 0.f;
    for (int i = i0; i < i1; ++i) {
      float ev = __expf(sc[m * kT + i] - mrow);
      sc[m * kT + i] = ev;
      sum += ev;
    }
    psum[seg][m] = sum;
    __syncthreads();
    if (tid < 16) {
      float v = 0.f;
      for (int s = 0; s < 8; ++s) v += psum[s][tid];
      rinv[tid] = 1.f / v;
    }
    __syncthreads();
    const float inv = rinv[m];
    for (int i = i0; i < i1; ++i) pb[m * 400 + i] = (__bf16)(sc[m * kT + i] * inv);
    if (tid < 16 && (L & 31))
      for (int j2 = 0; j2 < 16; ++j2) pb[tid * 400 + L + j2] = (__bf16)0.f;  // pad to K=32
  }
  __syncthreads();

  const int Lp = (L + 31) & ~31;
  const int jd = w;  // one d-tile per wave
  f32x8 oacc = zero8();
  for (int k0 = 0; k0 < Lp; k0 += 32) {
    bf16x16 af = load_a_frag(pb, 400, k0, lane);
    oacc = wmma_bf16(af, load_b_frag(VT + (size_t)(16 * jd) * kT, kT, k0, lane), oacc);
  }
  for (int r = 0; r < 8; ++r) {
    int m = r + mo;
    yb[((size_t)b * kT + t0 + m) * kC + h * kHD + 16 * jd + n] = (__bf16)oacc[r];
  }
}

// ---------------- WO projection + residual (fp32 out into d_out) ----------------
// grid (4 c-quads, 1536 token-tiles); 1 wave computes 16x64 with A reuse.

__global__ __launch_bounds__(32) void k_wo_res(const __bf16* __restrict__ yb,
                                               const __bf16* __restrict__ wob,
                                               const float* __restrict__ x,
                                               float* __restrict__ out) {
  int lane = threadIdx.x & 31;
  int ntq = blockIdx.x, tt = blockIdx.y;
  const __bf16* A  = yb  + (size_t)tt * 16 * kC;
  const __bf16* Bp = wob + (size_t)ntq * 64 * kC;
  f32x8 acc[4] = {zero8(), zero8(), zero8(), zero8()};
  for (int k0 = 0; k0 < kC; k0 += 32) {
    bf16x16 af = load_a_frag(A, kC, k0, lane);
    for (int j = 0; j < 4; ++j)
      acc[j] = wmma_bf16(af, load_b_frag(Bp + (size_t)16 * j * kC, kC, k0, lane), acc[j]);
  }
  const int n = lane & 15, mo = 8 * (lane >> 4);
  for (int j = 0; j < 4; ++j)
    for (int r = 0; r < 8; ++r) {
      size_t o = ((size_t)tt * 16 + r + mo) * kC + ntq * 64 + 16 * j + n;
      out[o] = x[o] + acc[j][r];
    }
}

// ---------------- RMSNorm2 + router softmax + top-2 gather ----------------

__global__ __launch_bounds__(32) void k_router(const float* __restrict__ xo,
                                               const float* __restrict__ w,
                                               const float* __restrict__ rw,
                                               __bf16* __restrict__ h2b,
                                               float* __restrict__ wts,
                                               int* __restrict__ cnt,
                                               int* __restrict__ idxb) {
  __shared__ float hsh[kC];
  __shared__ float lg[kNE];
  int row = blockIdx.x;
  int lane = threadIdx.x & 31;
  const float* xr = xo + (size_t)row * kC;
  float ss = 0.f;
  for (int i = lane; i < kC; i += 32) { float v = xr[i]; ss += v * v; }
  for (int off = 16; off; off >>= 1) ss += __shfl_xor(ss, off, 32);
  float r = rsqrtf(ss / (float)kC + 1e-6f);
  for (int i = lane; i < kC; i += 32) {
    float hv = xr[i] * r * w[i];
    hsh[i] = hv;
    h2b[(size_t)row * kC + i] = (__bf16)hv;
  }
  __syncthreads();
  if (lane < kNE) {
    float d = 0.f;
    for (int c = 0; c < kC; ++c) d += hsh[c] * rw[lane * kC + c];
    lg[lane] = d;
  }
  __syncthreads();
  if (lane == 0) {
    float mx = lg[0];
    for (int e = 1; e < kNE; ++e) mx = fmaxf(mx, lg[e]);
    float p[kNE], s = 0.f;
    for (int e = 0; e < kNE; ++e) { p[e] = __expf(lg[e] - mx); s += p[e]; }
    float inv = 1.f / s;
    for (int e = 0; e < kNE; ++e) p[e] *= inv;
    int i1 = 0;
    for (int e = 1; e < kNE; ++e) if (p[e] > p[i1]) i1 = e;
    int i2 = (i1 == 0) ? 1 : 0;
    for (int e = 0; e < kNE; ++e) if (e != i1 && p[e] > p[i2]) i2 = e;
    float s2 = p[i1] + p[i2] + 1e-9f;
    float w1v = p[i1] / s2, w2v = p[i2] / s2;
    for (int e = 0; e < kNE; ++e)
      wts[(size_t)row * kNE + e] = (e == i1) ? w1v : ((e == i2) ? w2v : 0.f);
    int p1 = atomicAdd(&cnt[i1], 1); idxb[i1 * kN + p1] = row;
    int p2 = atomicAdd(&cnt[i2], 1); idxb[i2 * kN + p2] = row;
  }
}

// ---------------- MoE expert kernel (TDM gather, WMMA, fused SwiGLU) ----------------
// grid (1536 tiles, NE); 128 threads (4 waves) per block.

__global__ __launch_bounds__(128) void k_moe(const __bf16* __restrict__ h2b,
                                             const __bf16* __restrict__ w1p,
                                             const __bf16* __restrict__ w3p,
                                             const __bf16* __restrict__ w2p,
                                             const float* __restrict__ wts,
                                             const int* __restrict__ cnt,
                                             const int* __restrict__ idxb,
                                             float* __restrict__ out) {
  __shared__ __align__(16) __bf16 xs[16 * kC];     // gathered token tile
  __shared__ __align__(16) __bf16 gb[16 * kHIDP];  // silu(h@w1^T)*(h@w3^T)
  __shared__ int   toks[16];
  __shared__ float tw[16];

  const int e = blockIdx.y;
  const int tile0 = blockIdx.x * 16;
  const int ne = cnt[e];
  if (tile0 >= ne) return;

  const int tid = threadIdx.x;

  // Wave-uniform scalar gather of the 16 token ids (s_load path).
  unsigned tokv[16];
#pragma unroll
  for (int j = 0; j < 16; ++j) {
    int i = tile0 + j;
    int ii = (i < ne) ? i : (ne - 1);
    int tok = idxb[e * kN + ii];
    tokv[j] = (unsigned)tok;
    if (tid == 0) {
      toks[j] = tok;
      tw[j] = (i < ne) ? wts[(size_t)tok * kNE + e] : 0.f;
    }
  }

#if HAS_TDM
  if (tid < 32) {
    // D# gather-mode descriptor (ISA 8.3-8.7): 16 rows of 256 x 2B from h2b -> xs
    u32x4 g0;
    unsigned ldsoff = (unsigned)(size_t)(&xs[0]);
    unsigned long long ga = (unsigned long long)(size_t)h2b;
    g0[0] = 0x80000001u;                                  // count=1, gather_mode=1, 16b idx
    g0[1] = ldsoff;                                       // lds_addr
    g0[2] = (unsigned)(ga & 0xFFFFFFFFu);                 // global_addr[31:0]
    g0[3] = (unsigned)((ga >> 32) & 0x1FFFFFFu) | 0x80000000u;  // [56:32] | type=2
    i32x8 g1;
    g1[0] = 0x00010000;       // data_size=1 (2 bytes)
    g1[1] = 0x01000000;       // tensor_dim0 = 256 (low16 @ bit48)
    g1[2] = 0x60000000;       // tensor_dim1 = 24576 (low16 @ bit80)
    g1[3] = 0x01000000;       // tile_dim0 = 256 (@ bit112)
    g1[4] = 16;               // tile_dim1 = #valid indices
    g1[5] = 256;              // tensor_dim0_stride
    g1[6] = 0;  g1[7] = 0;
    i32x4 g2, g3;
    g2[0] = (int)((tokv[0]  & 0xFFFFu) | (tokv[1]  << 16));
    g2[1] = (int)((tokv[2]  & 0xFFFFu) | (tokv[3]  << 16));
    g2[2] = (int)((tokv[4]  & 0xFFFFu) | (tokv[5]  << 16));
    g2[3] = (int)((tokv[6]  & 0xFFFFu) | (tokv[7]  << 16));
    g3[0] = (int)((tokv[8]  & 0xFFFFu) | (tokv[9]  << 16));
    g3[1] = (int)((tokv[10] & 0xFFFFu) | (tokv[11] << 16));
    g3[2] = (int)((tokv[12] & 0xFFFFu) | (tokv[13] << 16));
    g3[3] = (int)((tokv[14] & 0xFFFFu) | (tokv[15] << 16));
#if __clang_major__ >= 23
    i32x8 gz = {0,0,0,0,0,0,0,0};
    __builtin_amdgcn_tensor_load_to_lds(g0, g1, g2, g3, gz, 0);
#else
    __builtin_amdgcn_tensor_load_to_lds(g0, g1, g2, g3, 0);
#endif
    __builtin_amdgcn_s_wait_tensorcnt(0);
  }
  __syncthreads();
#else
  __syncthreads();
  for (int i = tid; i < 16 * (kC / 2); i += 128) {
    int row = i >> 7, c2 = i & 127;
    const uint32_t* src = (const uint32_t*)(h2b + (size_t)toks[row] * kC);
    ((uint32_t*)xs)[row * (kC / 2) + c2] = src[c2];
  }
  __syncthreads();
#endif

  const int w = tid >> 5, lane = tid & 31;
  const int n = lane & 15, mo = 8 * (lane >> 4);

  // GEMM1: [16 x HIDP] = xs @ {w1,w3}^T, A reused across 2 n-tiles x {w1,w3}
  const int nt_lo = 11 * w, nt_hi = nt_lo + 11;  // 44 tiles / 4 waves
  for (int ntb = nt_lo; ntb < nt_hi; ntb += 2) {
    const bool two = (ntb + 1) < nt_hi;
    const __bf16* B1a = w1p + ((size_t)e * kHIDP + ntb * 16) * kC;
    const __bf16* B3a = w3p + ((size_t)e * kHIDP + ntb * 16) * kC;
    f32x8 a1a = zero8(), a3a = zero8(), a1b = zero8(), a3b = zero8();
    for (int k0 = 0; k0 < kC; k0 += 32) {
      bf16x16 af = load_a_frag(xs, kC, k0, lane);
      a1a = wmma_bf16(af, load_b_frag(B1a, kC, k0, lane), a1a);
      a3a = wmma_bf16(af, load_b_frag(B3a, kC, k0, lane), a3a);
      if (two) {
        a1b = wmma_bf16(af, load_b_frag(B1a + (size_t)16 * kC, kC, k0, lane), a1b);
        a3b = wmma_bf16(af, load_b_frag(B3a + (size_t)16 * kC, kC, k0, lane), a3b);
      }
    }
    for (int r = 0; r < 8; ++r) {
      int m = r + mo;
      float g1v = a1a[r], g3v = a3a[r];
      gb[m * kHIDP + ntb * 16 + n] = (__bf16)((g1v / (1.f + __expf(-g1v))) * g3v);
    }
    if (two)
      for (int r = 0; r < 8; ++r) {
        int m = r + mo;
        float g1v = a1b[r], g3v = a3b[r];
        gb[m * kHIDP + (ntb + 1) * 16 + n] = (__bf16)((g1v / (1.f + __expf(-g1v))) * g3v);
      }
  }
  __syncthreads();

  // GEMM2: [16 x C] = gb @ w2^T; each wave owns 4 c-tiles, A reused 4x
  f32x8 acc[4] = {zero8(), zero8(), zero8(), zero8()};
  const __bf16* B2 = w2p + ((size_t)e * kC + w * 64) * kHIDP;
  for (int k0 = 0; k0 < kHIDP; k0 += 32) {
    bf16x16 af = load_a_frag(gb, kHIDP, k0, lane);
    for (int j = 0; j < 4; ++j)
      acc[j] = wmma_bf16(af, load_b_frag(B2 + (size_t)16 * j * kHIDP, kHIDP, k0, lane), acc[j]);
  }
  for (int j = 0; j < 4; ++j)
    for (int r = 0; r < 8; ++r) {
      int m = r + mo;
      float v = acc[j][r] * tw[m];
      __hip_atomic_fetch_add(&out[(size_t)toks[m] * kC + w * 64 + 16 * j + n], v,
                             __ATOMIC_RELAXED, __HIP_MEMORY_SCOPE_AGENT);
    }
}

// ---------------- Host launcher ----------------

extern "C" void kernel_launch(void* const* d_in, const int* in_sizes, int n_in,
                              void* d_out, int out_size, void* d_ws, size_t ws_size,
                              hipStream_t stream) {
  const float* x   = (const float*)d_in[0];
  const float* ln1 = (const float*)d_in[1];
  const float* ln2 = (const float*)d_in[2];
  const float* wq  = (const float*)d_in[3];
  const float* wk  = (const float*)d_in[4];
  const float* wv  = (const float*)d_in[5];
  const float* wo  = (const float*)d_in[6];
  const float* rw  = (const float*)d_in[7];
  const float* w1  = (const float*)d_in[8];
  const float* w2  = (const float*)d_in[9];
  const float* w3  = (const float*)d_in[10];
  float* out = (float*)d_out;

  char* p = (char*)d_ws;
  auto alloc = [&](size_t bytes) -> char* {
    char* r = p;
    p += (bytes + 255) & ~(size_t)255;
    return r;
  };

  __bf16* wqb = (__bf16*)alloc((size_t)kC * kC * 2);
  __bf16* wkb = (__bf16*)alloc((size_t)kNKV * kHD * kC * 2);
  __bf16* wvb = (__bf16*)alloc((size_t)kNKV * kHD * kC * 2);
  __bf16* wob = (__bf16*)alloc((size_t)kC * kC * 2);
  __bf16* w1p = (__bf16*)alloc((size_t)kNE * kHIDP * kC * 2);
  __bf16* w3p = (__bf16*)alloc((size_t)kNE * kHIDP * kC * 2);
  __bf16* w2p = (__bf16*)alloc((size_t)kNE * kC * kHIDP * 2);
  float*  rc  = (float*)alloc((size_t)kT * kHD * 4);
  float*  rs  = (float*)alloc((size_t)kT * kHD * 4);
  __bf16* hb  = (__bf16*)alloc((size_t)kN * kC * 2);
  __bf16* qb  = (__bf16*)alloc((size_t)kN * kC * 2);
  __bf16* kb2 = (__bf16*)alloc((size_t)kN * kNKV * kHD * 2);  // [B][KV][T][HD]
  __bf16* vtb = (__bf16*)alloc((size_t)kN * kNKV * kHD * 2);  // [B][KV][HD][T]
  __bf16* yb  = (__bf16*)alloc((size_t)kN * kC * 2);
  __bf16* h2b = (__bf16*)alloc((size_t)kN * kC * 2);
  float*  wts = (float*)alloc((size_t)kN * kNE * 4);
  int*    cnt = (int*)alloc(kNE * 4);
  int*    idxb = (int*)alloc((size_t)kNE * kN * 4);
  (void)ws_size; (void)in_sizes; (void)n_in; (void)out_size;

  // prep
  k_zero_cnt<<<1, 32, 0, stream>>>(cnt);
  k_cvt_bf16<<<(kC * kC + 255) / 256, 256, 0, stream>>>(wqb, wq, kC * kC);
  k_cvt_bf16<<<(kNKV * kHD * kC + 255) / 256, 256, 0, stream>>>(wkb, wk, kNKV * kHD * kC);
  k_cvt_bf16<<<(kNKV * kHD * kC + 255) / 256, 256, 0, stream>>>(wvb, wv, kNKV * kHD * kC);
  k_cvt_bf16<<<(kC * kC + 255) / 256, 256, 0, stream>>>(wob, wo, kC * kC);
  k_pad_w13<<<(kNE * kHIDP * kC + 255) / 256, 256, 0, stream>>>(w1p, w1);
  k_pad_w13<<<(kNE * kHIDP * kC + 255) / 256, 256, 0, stream>>>(w3p, w3);
  k_pad_w2<<<(kNE * kC * kHIDP + 255) / 256, 256, 0, stream>>>(w2p, w2);
  k_rope_tables<<<(kT * kHD + 255) / 256, 256, 0, stream>>>(rc, rs);

  // attention block
  k_rmsnorm1<<<kN, 32, 0, stream>>>(x, ln1, hb);
  k_qkv_rope<<<dim3(kT / 16, 8, kB), 32, 0, stream>>>(hb, wqb, wkb, wvb, rc, rs, qb, kb2, vtb);
  k_attn<<<dim3(kT / 16, kNH, kB), 128, 0, stream>>>(qb, kb2, vtb, yb);
  k_wo_res<<<dim3(4, kN / 16), 32, 0, stream>>>(yb, wob, x, out);

  // MoE block
  k_router<<<kN, 32, 0, stream>>>(out, ln2, rw, h2b, wts, cnt, idxb);
  k_moe<<<dim3(kN / 16, kNE), 128, 0, stream>>>(h2b, w1p, w3p, w2p, wts, cnt, idxb, out);
}
